// GCN_GC_29394756174083
// MI455X (gfx1250) — matrix operand order repaired and checked by reading, not compiled, not run on hardware
//
#include <hip/hip_runtime.h>
#include <hip/hip_bf16.h>

typedef unsigned short u16;
typedef __attribute__((ext_vector_type(16))) _Float16 v16h;
typedef __attribute__((ext_vector_type(8)))  float    v8f;

#define N_NODES 50000
#define N_FEAT  128
#define NUM_GRAPHS 128
#define OUT_CH  10

__device__ __forceinline__ u16 f2h(float f) {
    union { _Float16 h; u16 u; } v;
    v.h = (_Float16)f;
    return v.u;
}

// ---------------- utility -------------------------------------------------
__global__ void zero_f32(float* __restrict__ p, int n) {
    int i = blockIdx.x * blockDim.x + threadIdx.x;
    if (i < n) p[i] = 0.0f;
}

// deg[dst] += 1 over all edges
__global__ void deg_kernel(const long long* __restrict__ dst, float* __restrict__ deg, int ne) {
    int e = blockIdx.x * blockDim.x + threadIdx.x;
    if (e < ne) atomicAdd(&deg[(int)dst[e]], 1.0f);
}

// one wave per edge; lane handles 4 contiguous features (float4 gather + 4 f32 atomics)
__global__ void scatter_kernel(const float* __restrict__ h,
                               const long long* __restrict__ src,
                               const long long* __restrict__ dst,
                               float* __restrict__ agg, int ne) {
    int t = blockIdx.x * blockDim.x + threadIdx.x;
    int e = t >> 5;
    if (e >= ne) return;
    int lane = t & 31;
    int s = (int)src[e];
    int d = (int)dst[e];
    float4 v = *(const float4*)(h + (size_t)s * N_FEAT + lane * 4);
    float* ap = agg + (size_t)d * N_FEAT + lane * 4;
    atomicAdd(ap + 0, v.x);
    atomicAdd(ap + 1, v.y);
    atomicAdd(ap + 2, v.z);
    atomicAdd(ap + 3, v.w);
}

// agg_f16 = (agg / max(deg,1)) in f16 ; h_f16 = h in f16
__global__ void finalize_kernel(const float* __restrict__ agg, const float* __restrict__ deg,
                                const float* __restrict__ h,
                                u16* __restrict__ aggh, u16* __restrict__ hh, int n) {
    int i = blockIdx.x * blockDim.x + threadIdx.x;
    if (i >= n) return;
    int node = i >> 7;
    float d = deg[node];
    d = d < 1.0f ? 1.0f : d;
    aggh[i] = f2h(agg[i] / d);
    hh[i]   = f2h(h[i]);
}

// convert both weight matrices fp32 -> f16
__global__ void wconv_kernel(const float* __restrict__ Wl, const float* __restrict__ Wr,
                             u16* __restrict__ wlh, u16* __restrict__ wrh, int n) {
    int i = blockIdx.x * blockDim.x + threadIdx.x;
    if (i >= n) return;
    wlh[i] = f2h(Wl[i]);
    wrh[i] = f2h(Wr[i]);
}

// ---------------- fused SAGE GEMM: out = relu(agg@Wl^T + bias + h@Wr^T) ---
// Block: 256 threads = 8 waves. Wave w owns row-tile (blockIdx.x*8 + w) of 16 rows.
// Weights (both) staged in LDS (64KB of the WGP's 320KB).
union AF { uint4 q[2]; v16h h; };

__global__ __launch_bounds__(256)
void sage_gemm(const u16* __restrict__ aggh, const u16* __restrict__ hh,
               const u16* __restrict__ wl, const u16* __restrict__ wr,
               const float* __restrict__ bias, float* __restrict__ out, int mtiles) {
    __shared__ u16 ldsW[2 * 128 * 128];   // Wl @ 0, Wr @ 16384 (u16 units)

    { // stage weights: 8192 x uint4 total
        uint4* l4 = (uint4*)ldsW;
        const uint4* s0 = (const uint4*)wl;
        const uint4* s1 = (const uint4*)wr;
        for (int i = threadIdx.x; i < 4096; i += 256) {
            l4[i]        = s0[i];
            l4[i + 4096] = s1[i];
        }
    }
    __syncthreads();

    const int wave = threadIdx.x >> 5;
    const int lane = threadIdx.x & 31;
    const int tile = blockIdx.x * 8 + wave;
    if (tile >= mtiles) return;          // wave-uniform (50000 % 16 == 0)

    const int half = lane >> 4;          // K-half selector for A/B fragments
    const int l16  = lane & 15;          // A: row-in-tile (M) ; B/C/D: column (N)
    const int row  = tile * 16 + l16;

    // A fragments for all 4 K-blocks, both matrices, kept in VGPRs.
    // A 16x32 in this code's contiguous k order (B matches): lane holds 16 contiguous f16:
    //   K in [kb*32 + half*16, +16)
    AF Aa[4], Ah[4];
    const u16* abase = aggh + (size_t)row * N_FEAT;
    const u16* hbase = hh   + (size_t)row * N_FEAT;
#pragma unroll
    for (int kb = 0; kb < 4; ++kb) {
        int o1 = kb * 32 + half * 16;
        int o2 = o1 + 8;
        Aa[kb].q[0] = *(const uint4*)(abase + o1);
        Aa[kb].q[1] = *(const uint4*)(abase + o2);
        Ah[kb].q[0] = *(const uint4*)(hbase + o1);
        Ah[kb].q[1] = *(const uint4*)(hbase + o2);
    }

#pragma unroll
    for (int ct = 0; ct < 8; ++ct) {
        const int ncol = ct * 16 + l16;
        v8f acc = {};
        // B 32x16 in the same contiguous k order as A: lane = column N, 16 contiguous K values
        // (K = kb*32 + half*16 .. +15) of row-major W[n][k].
#pragma unroll
        for (int kb = 0; kb < 4; ++kb) {
            AF Bl, Br;
            const u16* pl = &ldsW[ncol * 128 + kb * 32 + half * 16];
            const u16* pr = &ldsW[128 * 128 + ncol * 128 + kb * 32 + half * 16];
            Bl.q[0] = *(const uint4*)(pl);
            Bl.q[1] = *(const uint4*)(pl + 8);
            Br.q[0] = *(const uint4*)(pr);
            Br.q[1] = *(const uint4*)(pr + 8);
            acc = __builtin_amdgcn_wmma_f32_16x16x32_f16(
                      false, Aa[kb].h, false, Bl.h, (short)0, acc, false, false);
            acc = __builtin_amdgcn_wmma_f32_16x16x32_f16(
                      false, Ah[kb].h, false, Br.h, (short)0, acc, false, false);
        }
        // C/D layout: VGPR r -> M = r + 8*half ; N = lane&15
        const float bv = bias[ncol];
        const int rbase = tile * 16 + 8 * half;
#pragma unroll
        for (int r = 0; r < 8; ++r) {
            float v = acc[r] + bv;
            out[(size_t)(rbase + r) * N_FEAT + ncol] = v > 0.0f ? v : 0.0f;
        }
    }
}

// ---------------- pooling + classifier ------------------------------------
__global__ void pool_sum_kernel(const float* __restrict__ h, const long long* __restrict__ batch,
                                float* __restrict__ gsum, int n) {
    int i = blockIdx.x * blockDim.x + threadIdx.x;
    if (i >= n) return;
    int node = i >> 7, f = i & 127;
    atomicAdd(&gsum[(int)batch[node] * N_FEAT + f], h[i]);
}

__global__ void pool_cnt_kernel(const long long* __restrict__ batch, float* __restrict__ gcnt, int n) {
    int i = blockIdx.x * blockDim.x + threadIdx.x;
    if (i < n) atomicAdd(&gcnt[(int)batch[i]], 1.0f);
}

__global__ void final_linear_kernel(const float* __restrict__ gsum, const float* __restrict__ gcnt,
                                    const float* __restrict__ W, const float* __restrict__ b,
                                    float* __restrict__ out) {
    int i = blockIdx.x * blockDim.x + threadIdx.x;
    if (i >= NUM_GRAPHS * OUT_CH) return;
    int g = i / OUT_CH, c = i % OUT_CH;
    float cnt = gcnt[g];
    cnt = cnt < 1.0f ? 1.0f : cnt;
    float s = 0.0f;
    for (int k = 0; k < N_FEAT; ++k) s += gsum[g * N_FEAT + k] * W[c * N_FEAT + k];
    out[i] = s / cnt + b[c];
}

// ---------------- launch --------------------------------------------------
extern "C" void kernel_launch(void* const* d_in, const int* in_sizes, int n_in,
                              void* d_out, int out_size, void* d_ws, size_t ws_size,
                              hipStream_t stream) {
    const float*     x    = (const float*)d_in[0];
    const long long* eidx = (const long long*)d_in[1];
    const long long* batch = (const long long*)d_in[2];
    const float* Wl[3] = { (const float*)d_in[3], (const float*)d_in[6], (const float*)d_in[9]  };
    const float* bl[3] = { (const float*)d_in[4], (const float*)d_in[7], (const float*)d_in[10] };
    const float* Wr[3] = { (const float*)d_in[5], (const float*)d_in[8], (const float*)d_in[11] };
    const float* W_lin = (const float*)d_in[12];
    const float* b_lin = (const float*)d_in[13];
    float* out = (float*)d_out;

    const int ne = in_sizes[1] / 2;
    const long long* src = eidx;
    const long long* dst = eidx + ne;

    const int NF = N_NODES * N_FEAT;  // 6,400,000

    // workspace carve-up (256B aligned)
    char* w = (char*)d_ws;
    size_t off = 0;
    auto alloc = [&](size_t bytes) -> void* {
        off = (off + 255) & ~(size_t)255;
        void* p = w + off;
        off += bytes;
        return p;
    };
    float* agg   = (float*)alloc((size_t)NF * 4);
    float* hbuf  = (float*)alloc((size_t)NF * 4);
    u16*   aggh  = (u16*)  alloc((size_t)NF * 2);
    u16*   hh    = (u16*)  alloc((size_t)NF * 2);
    float* deg   = (float*)alloc((size_t)N_NODES * 4);
    u16*   wlh   = (u16*)  alloc(128 * 128 * 2);
    u16*   wrh   = (u16*)  alloc(128 * 128 * 2);
    float* gsum  = (float*)alloc((NUM_GRAPHS * N_FEAT + NUM_GRAPHS) * 4);
    float* gcnt  = gsum + NUM_GRAPHS * N_FEAT;
    (void)ws_size; (void)n_in; (void)out_size;

    // degree (same for all layers)
    zero_f32<<<(N_NODES + 255) / 256, 256, 0, stream>>>(deg, N_NODES);
    deg_kernel<<<(ne + 255) / 256, 256, 0, stream>>>(dst, deg, ne);

    const int mtiles = N_NODES / 16;                 // 3125
    const int gemm_blocks = (mtiles + 7) / 8;        // 391

    for (int l = 0; l < 3; ++l) {
        const float* hprev = (l == 0) ? x : hbuf;
        zero_f32<<<(NF + 255) / 256, 256, 0, stream>>>(agg, NF);
        scatter_kernel<<<(ne * 32 + 255) / 256, 256, 0, stream>>>(hprev, src, dst, agg, ne);
        finalize_kernel<<<(NF + 255) / 256, 256, 0, stream>>>(agg, deg, hprev, aggh, hh, NF);
        wconv_kernel<<<(128 * 128 + 255) / 256, 256, 0, stream>>>(Wl[l], Wr[l], wlh, wrh, 128 * 128);
        // GEMM only reads f16 staging buffers -> safe to overwrite hbuf
        sage_gemm<<<gemm_blocks, 256, 0, stream>>>(aggh, hh, wlh, wrh, bl[l], hbuf, mtiles);
    }

    zero_f32<<<(NUM_GRAPHS * N_FEAT + NUM_GRAPHS + 255) / 256, 256, 0, stream>>>(
        gsum, NUM_GRAPHS * N_FEAT + NUM_GRAPHS);
    pool_sum_kernel<<<(NF + 255) / 256, 256, 0, stream>>>(hbuf, batch, gsum, NF);
    pool_cnt_kernel<<<(N_NODES + 255) / 256, 256, 0, stream>>>(batch, gcnt, N_NODES);
    final_linear_kernel<<<(NUM_GRAPHS * OUT_CH + 255) / 256, 256, 0, stream>>>(
        gsum, gcnt, W_lin, b_lin, out);
}
